// GAT_50775103373975
// MI455X (gfx1250) — matrix operand order repaired and checked by reading, not compiled, not run on hardware
//
#include <hip/hip_runtime.h>
#include <hip/hip_bf16.h>
#include <math.h>

typedef __attribute__((ext_vector_type(16))) __bf16 v16bf;
typedef __attribute__((ext_vector_type(8)))  __bf16 v8bf;
typedef __attribute__((ext_vector_type(8)))  float  v8f;

#define CIN 256
#define NEG_SLOPE 0.2f

// ---------- helpers ----------
__device__ __forceinline__ unsigned short f2bf(float f) {
  unsigned u = __float_as_uint(f);
  u += 0x7FFFu + ((u >> 16) & 1u);           // round-to-nearest-even
  return (unsigned short)(u >> 16);
}
// order-preserving float <-> uint map so segment-max uses global_atomic_max_u32
__device__ __forceinline__ unsigned enc(float f) {
  unsigned u = __float_as_uint(f);
  return (u & 0x80000000u) ? ~u : (u | 0x80000000u);
}
__device__ __forceinline__ float dec(unsigned u) {
  unsigned b = (u & 0x80000000u) ? (u & 0x7FFFFFFFu) : ~u;
  return __uint_as_float(b);
}

// ---------- 0) init: zero out, reset segmax/denom, x->bf16, pack W -> B-fragment layout ----------
// Wt index: tid = ((gct*8 + ks)*32 + lane)*16 + e
//   holds W[k][n] with k = ks*32 + 16*(e>>3) + 8*(lane>>4) + (e&7), n = gct*16 + (lane&15)
// so a B fragment is 16 consecutive bf16 per lane -> one 32B coalesced load.
__global__ void __launch_bounds__(256) init_kernel(
    float* __restrict__ out, unsigned* __restrict__ segmax, float* __restrict__ denom,
    unsigned short* __restrict__ xb, const float* __restrict__ x,
    unsigned short* __restrict__ Wt, const float* __restrict__ W,
    int out_total, int nh_total, int wt_total) {
  int i = blockIdx.x * 256 + threadIdx.x;
  if (i < out_total) {
    out[i] = 0.f;
    xb[i] = f2bf(x[i]);                      // row-major bf16 copy of x
  }
  if (i < nh_total) { segmax[i] = 0u; denom[i] = 0.f; }
  if (i < wt_total) {
    int e = i & 15, lane = (i >> 4) & 31, ks = (i >> 9) & 7, gct = i >> 12;
    int k = ks * 32 + ((e >> 3) << 4) + ((lane >> 4) << 3) + (e & 7);
    int n = (gct << 4) + (lane & 15);
    Wt[i] = f2bf(W[k * CIN + n]);
  }
}

// ---------- 1) h = x @ W via v_wmma_f32_16x16x32_bf16 ----------
// wave -> 16 rows x 64 cols (4 accumulators), K loop 8 x 32. 8 waves/block = 128 rows.
// A fragment: two 16B bf16 loads per lane (no in-loop conversion); B: one 32B load per tile.
__global__ void __launch_bounds__(256) gemm_wmma_kernel(
    const unsigned short* __restrict__ xb, const unsigned short* __restrict__ Wt,
    float* __restrict__ h, int nrows) {
  const int lane = threadIdx.x & 31;
  const int wave = threadIdx.x >> 5;
  const int m0 = (blockIdx.x * 8 + wave) * 16;
  if (m0 >= nrows) return;                         // whole-wave exit: EXEC stays all-1s for WMMA
  const int n0   = blockIdx.y * 64;
  const int lmod = lane & 15;
  const int lhi  = lane >> 4;                      // half-wave select
  const unsigned short* xrow = xb + (size_t)(m0 + lmod) * CIN + lhi * 8;
  const v16bf* Bf = (const v16bf*)Wt;
  const int gct0 = blockIdx.y * 4;

  v8f acc0 = {}, acc1 = {}, acc2 = {}, acc3 = {};
  for (int ks = 0; ks < 8; ++ks) {
    const int k0 = ks * 32;
    __builtin_prefetch(xrow + k0 + 32, 0, 1);      // global_prefetch_b8 for next K slab
    // A fragment: lane holds row m0+lmod, K = {k0+8*lhi+0..7, k0+16+8*lhi+0..7}
    union { v16bf v; v8bf half[2]; } af;
    af.half[0] = *(const v8bf*)(xrow + k0);
    af.half[1] = *(const v8bf*)(xrow + k0 + 16);

    v16bf b0 = Bf[(size_t)((gct0 + 0) * 8 + ks) * 32 + lane];
    v16bf b1 = Bf[(size_t)((gct0 + 1) * 8 + ks) * 32 + lane];
    v16bf b2 = Bf[(size_t)((gct0 + 2) * 8 + ks) * 32 + lane];
    v16bf b3 = Bf[(size_t)((gct0 + 3) * 8 + ks) * 32 + lane];

    acc0 = __builtin_amdgcn_wmma_f32_16x16x32_bf16(false, af.v, false, b0, (short)0, acc0, false, false);
    acc1 = __builtin_amdgcn_wmma_f32_16x16x32_bf16(false, af.v, false, b1, (short)0, acc1, false, false);
    acc2 = __builtin_amdgcn_wmma_f32_16x16x32_bf16(false, af.v, false, b2, (short)0, acc2, false, false);
    acc3 = __builtin_amdgcn_wmma_f32_16x16x32_bf16(false, af.v, false, b3, (short)0, acc3, false, false);
  }
  // C/D layout: VGPR r, lanes0-15 -> row m0+r, lanes16-31 -> row m0+r+8; col = lmod (+16 per tile)
  float* hbase = h + (size_t)(m0 + 8 * lhi) * CIN + n0 + lmod;
  #pragma unroll
  for (int r = 0; r < 8; ++r) {
    hbase[(size_t)r * CIN + 0]  = acc0[r];
    hbase[(size_t)r * CIN + 16] = acc1[r];
    hbase[(size_t)r * CIN + 32] = acc2[r];
    hbase[(size_t)r * CIN + 48] = acc3[r];
  }
}

// ---------- 2) a_src/a_dst: one wave per (node, head), wave32 shuffle reduce ----------
__global__ void __launch_bounds__(256) att_kernel(
    const float* __restrict__ h, const float* __restrict__ att_src,
    const float* __restrict__ att_dst, float* __restrict__ a_src,
    float* __restrict__ a_dst, int n) {
  int w = (int)((blockIdx.x * 256 + threadIdx.x) >> 5);
  int lane = threadIdx.x & 31;
  int node = w >> 2, hd = w & 3;
  if (node >= n) return;
  const float* hrow = h + (size_t)node * CIN + hd * 64;
  float s = hrow[lane] * att_src[hd * 64 + lane] + hrow[32 + lane] * att_src[hd * 64 + 32 + lane];
  float d = hrow[lane] * att_dst[hd * 64 + lane] + hrow[32 + lane] * att_dst[hd * 64 + 32 + lane];
  #pragma unroll
  for (int off = 16; off; off >>= 1) {
    s += __shfl_xor(s, off, 32);
    d += __shfl_xor(d, off, 32);
  }
  if (lane == 0) { a_src[node * 4 + hd] = s; a_dst[node * 4 + hd] = d; }
}

// ---------- 3) segment max via hardware uint atomic max ----------
__global__ void __launch_bounds__(256) edge_max_kernel(
    const int* __restrict__ ei, const float* __restrict__ a_src,
    const float* __restrict__ a_dst, unsigned* __restrict__ segmax, int E_, int n) {
  int e = blockIdx.x * 256 + threadIdx.x;
  int total = E_ + n;
  if (e >= total) return;
  int s, d;
  if (e < E_) { s = ei[e]; d = ei[E_ + e]; } else { s = d = e - E_; }
  float4 as = *(const float4*)(a_src + (size_t)s * 4);
  float4 ad = *(const float4*)(a_dst + (size_t)d * 4);
  float l[4] = { as.x + ad.x, as.y + ad.y, as.z + ad.z, as.w + ad.w };
  #pragma unroll
  for (int hd = 0; hd < 4; ++hd) {
    float v = l[hd] >= 0.f ? l[hd] : NEG_SLOPE * l[hd];
    atomicMax(segmax + (size_t)d * 4 + hd, enc(v));
  }
}

// ---------- 4) denom = segment_sum(exp(l - max)) via global_atomic_add_f32 ----------
__global__ void __launch_bounds__(256) edge_sum_kernel(
    const int* __restrict__ ei, const float* __restrict__ a_src,
    const float* __restrict__ a_dst, const unsigned* __restrict__ segmax,
    float* __restrict__ denom, int E_, int n) {
  int e = blockIdx.x * 256 + threadIdx.x;
  int total = E_ + n;
  if (e >= total) return;
  int s, d;
  if (e < E_) { s = ei[e]; d = ei[E_ + e]; } else { s = d = e - E_; }
  float4 as = *(const float4*)(a_src + (size_t)s * 4);
  float4 ad = *(const float4*)(a_dst + (size_t)d * 4);
  float l[4] = { as.x + ad.x, as.y + ad.y, as.z + ad.z, as.w + ad.w };
  #pragma unroll
  for (int hd = 0; hd < 4; ++hd) {
    float v = l[hd] >= 0.f ? l[hd] : NEG_SLOPE * l[hd];
    float m = dec(segmax[(size_t)d * 4 + hd]);
    atomicAdd(denom + (size_t)d * 4 + hd, __expf(v - m));
  }
}

// ---------- 5) scatter: one wave per edge; lane = 8 channels, head = lane/8 ----------
__global__ void __launch_bounds__(256) scatter_kernel(
    const int* __restrict__ ei, const float* __restrict__ a_src,
    const float* __restrict__ a_dst, const unsigned* __restrict__ segmax,
    const float* __restrict__ denom, const float* __restrict__ h,
    float* __restrict__ out, int E_, int n) {
  int w = (int)((blockIdx.x * 256 + threadIdx.x) >> 5);
  int lane = threadIdx.x & 31;
  int total = E_ + n;
  if (w >= total) return;
  int s, d;
  if (w < E_) { s = ei[w]; d = ei[E_ + w]; } else { s = d = w - E_; }
  int hd = lane >> 3;
  float l = a_src[(size_t)s * 4 + hd] + a_dst[(size_t)d * 4 + hd];
  l = l >= 0.f ? l : NEG_SLOPE * l;
  float m = dec(segmax[(size_t)d * 4 + hd]);
  float alpha = __expf(l - m) / (denom[(size_t)d * 4 + hd] + 1e-16f);

  const float4* hs = (const float4*)(h + (size_t)s * CIN + lane * 8);
  float4 v0 = hs[0], v1 = hs[1];
  float* o = out + (size_t)d * CIN + lane * 8;
  atomicAdd(o + 0, alpha * v0.x); atomicAdd(o + 1, alpha * v0.y);
  atomicAdd(o + 2, alpha * v0.z); atomicAdd(o + 3, alpha * v0.w);
  atomicAdd(o + 4, alpha * v1.x); atomicAdd(o + 5, alpha * v1.y);
  atomicAdd(o + 6, alpha * v1.z); atomicAdd(o + 7, alpha * v1.w);
}

// ---------- 6) out = relu(out + bias) ----------
__global__ void __launch_bounds__(256) finalize_kernel(
    float* __restrict__ out, const float* __restrict__ bias, int total) {
  int i = blockIdx.x * 256 + threadIdx.x;
  if (i >= total) return;
  float v = out[i] + bias[i & (CIN - 1)];
  out[i] = v > 0.f ? v : 0.f;
}

extern "C" void kernel_launch(void* const* d_in, const int* in_sizes, int n_in,
                              void* d_out, int out_size, void* d_ws, size_t ws_size,
                              hipStream_t stream) {
  const float* x       = (const float*)d_in[0];
  const int*   ei      = (const int*)d_in[1];
  const float* W       = (const float*)d_in[2];
  const float* att_src = (const float*)d_in[3];
  const float* att_dst = (const float*)d_in[4];
  const float* bias    = (const float*)d_in[5];
  float* out = (float*)d_out;

  const int N = in_sizes[0] / CIN;   // 50000
  const int E = in_sizes[1] / 2;     // 800000
  const int NH = N * 4;
  const int WT_TOTAL = CIN * CIN;    // 65536 packed bf16 elems
  const int OUT_TOTAL = N * CIN;
  const int TOTAL_EDGES = E + N;     // with self loops

  // workspace layout (256B aligned slabs)
  char* ws = (char*)d_ws;
  size_t off = 0;
  auto al = [](size_t v) { return (v + 255) & ~(size_t)255; };
  float*          h      = (float*)(ws + off);          off = al(off + (size_t)OUT_TOTAL * 4);
  unsigned short* xb     = (unsigned short*)(ws + off); off = al(off + (size_t)OUT_TOTAL * 2);
  unsigned short* Wt     = (unsigned short*)(ws + off); off = al(off + (size_t)WT_TOTAL * 2);
  float*          a_src  = (float*)(ws + off);          off = al(off + (size_t)NH * 4);
  float*          a_dst  = (float*)(ws + off);          off = al(off + (size_t)NH * 4);
  unsigned*       segmax = (unsigned*)(ws + off);       off = al(off + (size_t)NH * 4);
  float*          denom  = (float*)(ws + off);          off = al(off + (size_t)NH * 4);

  // 0) init (zero accumulators, convert x to bf16, pack W)
  init_kernel<<<(OUT_TOTAL + 255) / 256, 256, 0, stream>>>(
      out, segmax, denom, xb, x, Wt, W, OUT_TOTAL, NH, WT_TOTAL);

  // 1) h = x @ W (bf16 WMMA, fp32 accumulate)
  dim3 ggrid((N / 16 + 7) / 8, 4);
  gemm_wmma_kernel<<<ggrid, 256, 0, stream>>>(xb, Wt, h, N);

  // 2) attention dots
  att_kernel<<<(NH + 7) / 8, 256, 0, stream>>>(h, att_src, att_dst, a_src, a_dst, N);

  // 3) segment max
  edge_max_kernel<<<(TOTAL_EDGES + 255) / 256, 256, 0, stream>>>(
      ei, a_src, a_dst, segmax, E, N);

  // 4) segment sum of exp
  edge_sum_kernel<<<(TOTAL_EDGES + 255) / 256, 256, 0, stream>>>(
      ei, a_src, a_dst, segmax, denom, E, N);

  // 5) weighted scatter-add (one wave per edge)
  scatter_kernel<<<(TOTAL_EDGES + 7) / 8, 256, 0, stream>>>(
      ei, a_src, a_dst, segmax, denom, h, out, E, N);

  // 6) bias + relu
  finalize_kernel<<<(OUT_TOTAL + 255) / 256, 256, 0, stream>>>(out, bias, OUT_TOTAL);
}
